// GptOssGroupedExperts_27118423507326
// MI455X (gfx1250) — matrix operand order repaired and checked by reading, not compiled, not run on hardware
//
#include <hip/hip_runtime.h>
#include <hip/hip_bf16.h>

// ---------------------------------------------------------------------------
// GPT-OSS grouped experts MLP on MI455X (gfx1250, wave32, WMMA)
//   GEMM1: [T,D] x W1e^T[D,2H] + b1 -> interleaved SwiGLU -> act [T,H] (ws)
//   GEMM2: [T,H] x W2e^T[H,D] + b2 -> out [T,D]
// fp32 data computed via bf16x3 split (hi*hi + hi*lo + lo*hi) on
// v_wmma_f32_16x16x32_bf16 -> ~2.7x the fp32-WMMA (16x16x4) FLOP/issue rate
// at fp32-class accuracy.
// ---------------------------------------------------------------------------

typedef __attribute__((ext_vector_type(16))) __bf16 v16bf;
typedef __attribute__((ext_vector_type(8)))  __bf16 v8bf;
typedef __attribute__((ext_vector_type(4)))  __bf16 v4bf;
typedef __attribute__((ext_vector_type(8)))  float  v8f;

#define DDIM   1024
#define HDIM   1024
#define TWO_H  2048
#define NEXP   32
#define BM     128
#define BN     128
#define BK     32
#define LDA    40        // padded LDS row stride (bf16 elems): 80B rows, 16B aligned, conflict-free
#define ALPHA  1.702f
#define LIMIT  7.0f

static __device__ __forceinline__ v16bf cat8(v8bf a, v8bf b) {
  return __builtin_shufflevector(a, b, 0,1,2,3,4,5,6,7,8,9,10,11,12,13,14,15);
}

// cooperative 128x32 fp32 tile -> bf16 hi/lo LDS tiles (row-major, LDA stride)
static __device__ __forceinline__ void load_tile_split(const float* __restrict__ src, int ldg,
                                                       __bf16* __restrict__ dhi,
                                                       __bf16* __restrict__ dlo, int tid) {
  const int r = tid >> 3;          // 0..31
  const int c = (tid & 7) << 2;    // 0,4,...,28
#pragma unroll
  for (int i = 0; i < 4; ++i) {
    const int row = r + (i << 5);
    const float4 v = *reinterpret_cast<const float4*>(src + (size_t)row * ldg + c);
    float f0 = v.x, f1 = v.y, f2 = v.z, f3 = v.w;
    v4bf hi, lo;
    __bf16 h;
    h = (__bf16)f0; hi[0] = h; lo[0] = (__bf16)(f0 - (float)h);
    h = (__bf16)f1; hi[1] = h; lo[1] = (__bf16)(f1 - (float)h);
    h = (__bf16)f2; hi[2] = h; lo[2] = (__bf16)(f2 - (float)h);
    h = (__bf16)f3; hi[3] = h; lo[3] = (__bf16)(f3 - (float)h);
    *reinterpret_cast<v4bf*>(dhi + row * LDA + c) = hi;
    *reinterpret_cast<v4bf*>(dlo + row * LDA + c) = lo;
  }
}

// A fragment (16x32 bf16): lane half h holds K runs [8h,8h+8) and [16+8h,24+8h)
static __device__ __forceinline__ v16bf fragA(const __bf16* base, int row, int kh8) {
  const __bf16* p = base + row * LDA;
  v8bf c0 = *reinterpret_cast<const v8bf*>(p + kh8);
  v8bf c1 = *reinterpret_cast<const v8bf*>(p + 16 + kh8);
  return cat8(c0, c1);
}

// B fragment (32x16 bf16): lane half h holds K runs [8h,8h+8) and [16+8h,24+8h) at col n=lane%16
static __device__ __forceinline__ v16bf fragB(const __bf16* base, int row, int kh16) {
  const __bf16* p = base + row * LDA + kh16;
  v8bf c0 = *reinterpret_cast<const v8bf*>(p);
  v8bf c1 = *reinterpret_cast<const v8bf*>(p + 16);
  return cat8(c0, c1);
}

static __device__ __forceinline__ int find_expert(const int* __restrict__ counts, long m0, long* start) {
  int e = 0; long s = 0;
  while (e < NEXP - 1) {
    int c = counts[e];
    if (s + c > m0) break;
    s += c; ++e;
  }
  *start = s;
  return e;
}

// ---------------------------------------------------------------------------
// Kernel 1: grouped GEMM1 + bias + interleaved SwiGLU -> act [T,H]
// ---------------------------------------------------------------------------
__global__ __launch_bounds__(256, 2)
void moe_gemm1_swiglu(const float* __restrict__ x, const int* __restrict__ counts,
                      const float* __restrict__ w1, const float* __restrict__ b1,
                      float* __restrict__ act) {
  const int tid  = threadIdx.x;
  const int lane = tid & 31;
  const int wv   = tid >> 5;
  const int wm   = (wv >> 2) << 6;   // 0 or 64
  const int wn   = (wv & 3) << 5;    // 0,32,64,96
  const int l16  = lane & 15;
  const int kh   = lane >> 4;        // 0/1

  const long m0 = (long)blockIdx.x * BM;
  const int  nb = blockIdx.y * BN;

  long seg_start;
  const int e = find_expert(counts, m0, &seg_start);
  const float* w1e = w1 + (size_t)e * TWO_H * DDIM;

  __shared__ __bf16 sm[4 * BM * LDA];
  __bf16* aHi = sm;
  __bf16* aLo = sm + BM * LDA;
  __bf16* bHi = sm + 2 * BM * LDA;
  __bf16* bLo = sm + 3 * BM * LDA;

  v8f acc[4][2] = {};

  const float* aSrc = x + (size_t)m0 * DDIM;
  const float* bSrc = w1e + (size_t)nb * DDIM;

  for (int kk = 0; kk < DDIM; kk += BK) {
    load_tile_split(aSrc + kk, DDIM, aHi, aLo, tid);
    load_tile_split(bSrc + kk, DDIM, bHi, bLo, tid);
    if (kk + BK < DDIM) {  // global_prefetch_b8 for the next K-chunk
      __builtin_prefetch(aSrc + kk + BK + (size_t)(tid >> 3) * DDIM + ((tid & 7) << 2), 0, 1);
      __builtin_prefetch(bSrc + kk + BK + (size_t)(tid >> 3) * DDIM + ((tid & 7) << 2), 0, 1);
    }
    __syncthreads();

    v16bf fBh[2], fBl[2];
#pragma unroll
    for (int nj = 0; nj < 2; ++nj) {
      const int brow = wn + (nj << 4) + l16;
      fBh[nj] = fragB(bHi, brow, kh << 3);
      fBl[nj] = fragB(bLo, brow, kh << 3);
    }
#pragma unroll
    for (int mi = 0; mi < 4; ++mi) {
      const int arow = wm + (mi << 4) + l16;
      v16bf fAh = fragA(aHi, arow, kh << 3);
      v16bf fAl = fragA(aLo, arow, kh << 3);
#pragma unroll
      for (int nj = 0; nj < 2; ++nj) {
        acc[mi][nj] = __builtin_amdgcn_wmma_f32_16x16x32_bf16(false, fAh, false, fBh[nj],
                                                              (short)0, acc[mi][nj], false, false);
        acc[mi][nj] = __builtin_amdgcn_wmma_f32_16x16x32_bf16(false, fAh, false, fBl[nj],
                                                              (short)0, acc[mi][nj], false, false);
        acc[mi][nj] = __builtin_amdgcn_wmma_f32_16x16x32_bf16(false, fAl, false, fBh[nj],
                                                              (short)0, acc[mi][nj], false, false);
      }
    }
    __syncthreads();
  }

  // epilogue: bias + interleaved SwiGLU; even/odd column pair sits in adjacent lanes
  float bias[2];
#pragma unroll
  for (int nj = 0; nj < 2; ++nj)
    bias[nj] = b1[(size_t)e * TWO_H + nb + wn + (nj << 4) + l16];

#pragma unroll
  for (int mi = 0; mi < 4; ++mi) {
#pragma unroll
    for (int nj = 0; nj < 2; ++nj) {
#pragma unroll
      for (int r = 0; r < 8; ++r) {
        float v = acc[mi][nj][r] + bias[nj];
        float p = __shfl_xor(v, 1, 32);          // partner column (odd<->even)
        if ((l16 & 1) == 0) {
          float g = fminf(v, LIMIT);
          float l = fminf(fmaxf(p, -LIMIT), LIMIT);
          float res = (g / (1.0f + __expf(-ALPHA * g))) * (l + 1.0f);
          const int t   = (int)m0 + wm + (mi << 4) + r + (kh << 3);
          const int col = nb + wn + (nj << 4) + l16;
          act[(size_t)t * HDIM + (col >> 1)] = res;
        }
      }
    }
  }
}

// ---------------------------------------------------------------------------
// Kernel 2: grouped GEMM2 + bias -> out [T,D]
// ---------------------------------------------------------------------------
__global__ __launch_bounds__(256, 2)
void moe_gemm2(const float* __restrict__ act, const int* __restrict__ counts,
               const float* __restrict__ w2, const float* __restrict__ b2,
               float* __restrict__ out) {
  const int tid  = threadIdx.x;
  const int lane = tid & 31;
  const int wv   = tid >> 5;
  const int wm   = (wv >> 2) << 6;
  const int wn   = (wv & 3) << 5;
  const int l16  = lane & 15;
  const int kh   = lane >> 4;

  const long m0 = (long)blockIdx.x * BM;
  const int  nb = blockIdx.y * BN;

  long seg_start;
  const int e = find_expert(counts, m0, &seg_start);
  const float* w2e = w2 + (size_t)e * DDIM * HDIM;

  __shared__ __bf16 sm[4 * BM * LDA];
  __bf16* aHi = sm;
  __bf16* aLo = sm + BM * LDA;
  __bf16* bHi = sm + 2 * BM * LDA;
  __bf16* bLo = sm + 3 * BM * LDA;

  v8f acc[4][2] = {};

  const float* aSrc = act + (size_t)m0 * HDIM;
  const float* bSrc = w2e + (size_t)nb * HDIM;

  for (int kk = 0; kk < HDIM; kk += BK) {
    load_tile_split(aSrc + kk, HDIM, aHi, aLo, tid);
    load_tile_split(bSrc + kk, HDIM, bHi, bLo, tid);
    if (kk + BK < HDIM) {
      __builtin_prefetch(aSrc + kk + BK + (size_t)(tid >> 3) * HDIM + ((tid & 7) << 2), 0, 1);
      __builtin_prefetch(bSrc + kk + BK + (size_t)(tid >> 3) * HDIM + ((tid & 7) << 2), 0, 1);
    }
    __syncthreads();

    v16bf fBh[2], fBl[2];
#pragma unroll
    for (int nj = 0; nj < 2; ++nj) {
      const int brow = wn + (nj << 4) + l16;
      fBh[nj] = fragB(bHi, brow, kh << 3);
      fBl[nj] = fragB(bLo, brow, kh << 3);
    }
#pragma unroll
    for (int mi = 0; mi < 4; ++mi) {
      const int arow = wm + (mi << 4) + l16;
      v16bf fAh = fragA(aHi, arow, kh << 3);
      v16bf fAl = fragA(aLo, arow, kh << 3);
#pragma unroll
      for (int nj = 0; nj < 2; ++nj) {
        acc[mi][nj] = __builtin_amdgcn_wmma_f32_16x16x32_bf16(false, fAh, false, fBh[nj],
                                                              (short)0, acc[mi][nj], false, false);
        acc[mi][nj] = __builtin_amdgcn_wmma_f32_16x16x32_bf16(false, fAh, false, fBl[nj],
                                                              (short)0, acc[mi][nj], false, false);
        acc[mi][nj] = __builtin_amdgcn_wmma_f32_16x16x32_bf16(false, fAl, false, fBh[nj],
                                                              (short)0, acc[mi][nj], false, false);
      }
    }
    __syncthreads();
  }

  float bias[2];
#pragma unroll
  for (int nj = 0; nj < 2; ++nj)
    bias[nj] = b2[(size_t)e * DDIM + nb + wn + (nj << 4) + l16];

#pragma unroll
  for (int mi = 0; mi < 4; ++mi) {
#pragma unroll
    for (int nj = 0; nj < 2; ++nj) {
      const int col = nb + wn + (nj << 4) + l16;
#pragma unroll
      for (int r = 0; r < 8; ++r) {
        const int t = (int)m0 + wm + (mi << 4) + r + (kh << 3);
        out[(size_t)t * DDIM + col] = acc[mi][nj][r] + bias[nj];
      }
    }
  }
}

// ---------------------------------------------------------------------------
extern "C" void kernel_launch(void* const* d_in, const int* in_sizes, int n_in,
                              void* d_out, int out_size, void* d_ws, size_t ws_size,
                              hipStream_t stream) {
  const float* x      = (const float*)d_in[0];
  const int*   counts = (const int*)  d_in[1];
  const float* w1     = (const float*)d_in[2];
  const float* b1     = (const float*)d_in[3];
  const float* w2     = (const float*)d_in[4];
  const float* b2     = (const float*)d_in[5];
  float*       out    = (float*)d_out;
  float*       act    = (float*)d_ws;   // [T, H] fp32 intermediate

  const int T = in_sizes[0] / DDIM;     // 65536
  dim3 blk(256);
  dim3 g1(T / BM, TWO_H / BN);          // 512 x 16
  dim3 g2(T / BM, DDIM / BN);           // 512 x 8
  moe_gemm1_swiglu<<<g1, blk, 0, stream>>>(x, counts, w1, b1, act);
  moe_gemm2<<<g2, blk, 0, stream>>>(act, counts, w2, b2, out);
}
